// DREAMCell_51608327029266
// MI455X (gfx1250) — matrix operand order repaired and checked by reading, not compiled, not run on hardware
//
#include <hip/hip_runtime.h>
#include <math.h>

// ---------------- problem constants (match reference) ----------------
#define B_TOT   32
#define T_STEPS 128
#define D_DIM   512
#define H_DIM   1024
#define R_DIM   16

// ---------------- decomposition ----------------
#define N_CL     2                  // clusters (batch groups of 16)
#define NWG_CL   16                 // workgroups per cluster (H split)
#define MB       16                 // batches per cluster  (WMMA M dim)
#define HS       (H_DIM / NWG_CL)   // 64 h-rows per WG
#define NTHREADS 256                // 8 wave32

typedef __bf16 bf16_t;
typedef bf16_t v16bf __attribute__((ext_vector_type(16)));
typedef bf16_t v8bf  __attribute__((ext_vector_type(8)));
typedef float  v8f   __attribute__((ext_vector_type(8)));

union Pk4 { bf16_t h[4]; uint2 u; };

// ---- 16-bit A fragment (16x32 MxK) from bf16 row-major, lane m=l&15 ----
// layout: vgpr0..3 <- K = kh*16 + 0..7 ; vgpr4..7 <- K = kh*16 + 8 + 0..7
__device__ __forceinline__ v16bf frag_A_bf(const bf16_t* src, int ld, int lane) {
  const int m  = lane & 15;
  const int kh = (lane >> 4) & 1;
  const bf16_t* p = src + m * ld + (kh << 4);
  const v8bf lo = *(const v8bf*)p;          // ds_load_b128
  const v8bf hi = *(const v8bf*)(p + 8);    // ds_load_b128
  v16bf a;
#pragma unroll
  for (int j = 0; j < 8; ++j) { a[j] = lo[j]; a[8 + j] = hi[j]; }
  return a;
}

// ---- 16-bit B fragment (32x16 KxN) from bf16, src[n*ld + k]: one 32B load ----
__device__ __forceinline__ v16bf frag_B_bf(const bf16_t* src, int ld, int lane) {
  const int n  = lane & 15;
  const int kh = (lane >> 4) & 1;
  return *(const v16bf*)(src + n * ld + (kh << 4));
}

__device__ __forceinline__ v8f wmma_bf16(v16bf a, v16bf b, v8f c) {
  return __builtin_amdgcn_wmma_f32_16x16x32_bf16(false, a, false, b, (short)0, c,
                                                 false, false);
}

// ---- C/D layout: lane n=l&15, vgpr r -> m = (l>>4)*8 + r ----
__device__ __forceinline__ void acc_to_lds(float* dst, int ld, v8f acc, int lane) {
  const int n  = lane & 15;
  const int mh = (lane >> 4) & 1;
#pragma unroll
  for (int r = 0; r < 8; ++r) dst[(mh * 8 + r) * ld + n] = acc[r];
}

// ---- cluster barrier: hw split-barrier (NOP w/o cluster launch) backed by a
// ---- cumulative agent-scope counter. Max inter-WG skew is 1 epoch, so
// ---- sum >= 16*epoch <=> every WG signaled this epoch.
__device__ __forceinline__ void cl_signal(unsigned* ctr) {
  __threadfence();                       // release: global_wb DEV + waits
  __syncthreads();
  if (threadIdx.x == 0)
    __hip_atomic_fetch_add(ctr, 1u, __ATOMIC_RELEASE, __HIP_MEMORY_SCOPE_AGENT);
  if (threadIdx.x < 32u) asm volatile("s_barrier_signal -3" ::: "memory");
}
__device__ __forceinline__ void cl_wait(unsigned* ctr, unsigned target) {
  asm volatile("s_barrier_wait -3" ::: "memory");
  if (threadIdx.x == 0) {
    while (__hip_atomic_load(ctr, __ATOMIC_ACQUIRE, __HIP_MEMORY_SCOPE_AGENT) < target)
      __builtin_amdgcn_s_sleep(1);       // acquire: global_inv -> WGP caches clean
  }
  __syncthreads();
}

__global__ __launch_bounds__(NTHREADS) void dream_cell_kernel(
    const float* __restrict__ x_seq,   // (B,T,D)
    const float* __restrict__ W_pred,  // (D,H)
    const float* __restrict__ W_in,    // (H,D)
    const float* __restrict__ W_err,   // (H,D)
    const float* __restrict__ Vm,      // (R,D)
    float* __restrict__ out,           // (B,T,H)
    unsigned* __restrict__ ws_ctr,
    float* __restrict__ ws_xp,         // [2][N_CL][NWG_CL][MB*D] scrambled partials
    float* __restrict__ ws_fro,        // [2][N_CL][NWG_CL][MB]
    bf16_t* __restrict__ wp_bf,        // (D,H)  bf16 weights (one-time convert)
    bf16_t* __restrict__ wi_bf,        // (H,D)
    bf16_t* __restrict__ we_bf,        // (H,D)
    bf16_t* __restrict__ v_bf)         // (R,D)
{
  extern __shared__ __align__(32) char smem_raw[];
  bf16_t* errbh = (bf16_t*)smem_raw;            // MB*D   bf16 error
  bf16_t* xbh   = errbh + MB * D_DIM;           // MB*D   bf16 x_t
  bf16_t* hbh   = xbh   + MB * D_DIM;           // MB*HS  bf16 h slice
  float* Ubuf   = (float*)(hbh + MB * HS);      // MB*HS*R
  float* Utgt   = Ubuf  + MB * HS * R_DIM;
  float* errb   = Utgt  + MB * HS * R_DIM;      // MB*D f32 error
  float* emean  = errb  + MB * D_DIM;
  float* evar   = emean + MB * D_DIM;
  float* hb     = evar  + MB * D_DIM;           // MB*HS f32 h
  float* baseb  = hb    + MB * HS;
  float* errwb  = baseb + MB * HS;
  float* eproj  = errwb + MB * HS;              // MB*R
  float* xproj  = eproj + MB * R_DIM;           // MB*R
  float* s_mu   = xproj + MB * R_DIM;
  float* s_sig  = s_mu  + MB;
  float* avg_s  = s_sig + MB;
  float* surp   = avg_s + MB;
  float* scl    = surp  + MB;

  const int tid  = threadIdx.x;
  const int lane = tid & 31;
  const int wid  = tid >> 5;
  const int cl   = blockIdx.x >> 4;    // cluster (batch group)
  const int cw   = blockIdx.x & 15;    // wg-in-cluster (H slice)
  const int h0   = cw * HS;
  const int bbase = cl * MB;
  unsigned* ctr = ws_ctr + cl * 32;    // 128B apart per cluster

  // ---- init recurrent state ----
  for (int i = tid; i < MB * HS * R_DIM; i += NTHREADS) { Ubuf[i] = 0.f; Utgt[i] = 0.f; }
  for (int i = tid; i < MB * D_DIM;     i += NTHREADS) { emean[i] = 0.f; evar[i] = 1.f; }
  for (int i = tid; i < MB * HS;        i += NTHREADS) { hb[i] = 0.f; hbh[i] = (bf16_t)0.f; }
  if (tid < MB) { s_mu[tid] = 0.f; s_sig[tid] = 1.f; avg_s[tid] = 0.f; }

  // ---- one-time weight f32->bf16 conversion (each cluster writes the full
  // ---- set; duplicate identical writes across clusters are benign) ----
  {
    const int start = (cw * NTHREADS + tid) * 4;
    const int step  = NWG_CL * NTHREADS * 4;
    for (int i = start; i < D_DIM * H_DIM; i += step) {
      float4 a = *(const float4*)&W_pred[i];
      Pk4 p; p.h[0]=(bf16_t)a.x; p.h[1]=(bf16_t)a.y; p.h[2]=(bf16_t)a.z; p.h[3]=(bf16_t)a.w;
      *(uint2*)&wp_bf[i] = p.u;
      a = *(const float4*)&W_in[i];
      p.h[0]=(bf16_t)a.x; p.h[1]=(bf16_t)a.y; p.h[2]=(bf16_t)a.z; p.h[3]=(bf16_t)a.w;
      *(uint2*)&wi_bf[i] = p.u;
      a = *(const float4*)&W_err[i];
      p.h[0]=(bf16_t)a.x; p.h[1]=(bf16_t)a.y; p.h[2]=(bf16_t)a.z; p.h[3]=(bf16_t)a.w;
      *(uint2*)&we_bf[i] = p.u;
    }
    for (int i = start; i < R_DIM * D_DIM; i += step) {
      float4 a = *(const float4*)&Vm[i];
      Pk4 p; p.h[0]=(bf16_t)a.x; p.h[1]=(bf16_t)a.y; p.h[2]=(bf16_t)a.z; p.h[3]=(bf16_t)a.w;
      *(uint2*)&v_bf[i] = p.u;
    }
  }
  unsigned epoch = 1;
  cl_signal(ctr); cl_wait(ctr, epoch * NWG_CL);

  for (int t = 0; t < T_STEPS; ++t) {
    const int par = t & 1;  // double-buffered exchange slots
    float*       xp_mine  = ws_xp + ((((size_t)par * N_CL + cl) * NWG_CL + cw) * MB) * D_DIM;
    const float* xp_base  = ws_xp + (((size_t)par * N_CL + cl) * NWG_CL) * MB * D_DIM;
    float*       fro_mine = ws_fro + (((size_t)par * N_CL + cl) * NWG_CL + cw) * MB;
    const float* fro_base = ws_fro + (((size_t)par * N_CL + cl) * NWG_CL) * MB;

    if (t + 1 < T_STEPS) {  // prefetch next x_t tile (32KB, 128B/thread)
      const int pm = tid >> 4, pd = (tid & 15) << 5;
      __builtin_prefetch(&x_seq[((size_t)(bbase + pm) * T_STEPS + (t + 1)) * D_DIM + pd], 0, 1);
    }

    // ---- Phase A: x_pred partial = h_slice(16x64) @ W_pred[:,h-slice]^T ----
    // Partials stored in WMMA-native "scrambled" layout so each lane emits one
    // contiguous 32B store:  p = nt*256 + lane*8 + r
    //   <->  d = nt*16 + (lane&15),  m = (lane>>4)*8 + r
    {
      const v16bf a0 = frag_A_bf(hbh, HS, lane);        // K 0..31 of slice
      const v16bf a1 = frag_A_bf(hbh + 32, HS, lane);   // K 32..63
#pragma unroll 2
      for (int i = 0; i < 4; ++i) {
        const int nt = wid * 4 + i;                     // 32 n-tiles over D
        const bf16_t* wb = wp_bf + (size_t)(nt * 16) * H_DIM + h0;
        v8f acc0 = {}, acc1 = {};
        acc0 = wmma_bf16(a0, frag_B_bf(wb,      H_DIM, lane), acc0);
        acc1 = wmma_bf16(a1, frag_B_bf(wb + 32, H_DIM, lane), acc1);
        *(v8f*)&xp_mine[(nt * 32 + lane) << 3] = acc0 + acc1;   // 2x b128
      }
    }
    ++epoch; cl_signal(ctr); cl_wait(ctr, epoch * NWG_CL);

    // ---- Phase B1: reduce scrambled partials across cluster, un-permute into
    // ---- errb[m][d] + bf16 shadows ----
    for (int p8 = tid * 8; p8 < MB * D_DIM; p8 += NTHREADS * 8) {
      v8f s = {};
#pragma unroll
      for (int w = 0; w < NWG_CL; ++w)
        s += *(const v8f*)&xp_base[(size_t)w * MB * D_DIM + p8];
      const int l = (p8 >> 3) & 31;
      const int d = ((p8 >> 8) << 4) + (l & 15);
      const int mb = (l >> 4) << 3;
#pragma unroll
      for (int r = 0; r < 8; ++r) {
        const int m = mb + r;
        const float xv = x_seq[((size_t)(bbase + m) * T_STEPS + t) * D_DIM + d];
        const float e  = xv - s[r];
        errb[m * D_DIM + d]  = e;
        errbh[m * D_DIM + d] = (bf16_t)e;
        xbh[m * D_DIM + d]   = (bf16_t)xv;
      }
    }
    __syncthreads();

    // ---- Phase B2: surprise gate stats (half-wave per batch) ----
    {
      const int m  = wid * 2 + (lane >> 4);
      const int li = lane & 15;
      float sz2 = 0.f, amax = -1e30f;
      for (int d = li; d < D_DIM; d += 16) {
        const float e = errb[m * D_DIM + d];
        const float z = (e - emean[m * D_DIM + d]) * rsqrtf(evar[m * D_DIM + d] + 1e-6f);
        sz2 += z * z;
        amax = fmaxf(amax, fabsf(e));
      }
#pragma unroll
      for (int msk = 1; msk < 16; msk <<= 1) {
        sz2 += __shfl_xor(sz2, msk, 16);
        amax = fmaxf(amax, __shfl_xor(amax, msk, 16));
      }
      float sume = 0.f;
      for (int d = li; d < D_DIM; d += 16)
        sume += __expf(fabsf(errb[m * D_DIM + d]) - amax);
#pragma unroll
      for (int msk = 1; msk < 16; msk <<= 1) sume += __shfl_xor(sume, msk, 16);
      const float inv = 1.f / sume;
      float ent = 0.f;
      for (int d = li; d < D_DIM; d += 16) {
        const float p = __expf(fabsf(errb[m * D_DIM + d]) - amax) * inv;
        ent -= p * __logf(p + 1e-6f);
      }
#pragma unroll
      for (int msk = 1; msk < 16; msk <<= 1) ent += __shfl_xor(ent, msk, 16);
      if (li == 0) {
        const float raw   = sz2 * (1.f / D_DIM);
        const float smun  = 0.99f * s_mu[m] + 0.01f * raw;
        const float ssign = 0.99f * s_sig[m] + 0.01f * fabsf(raw - smun);
        const float snorm = (raw - smun) / (ssign + 1e-6f);
        const float thr   = 0.5f + 0.1f * ent;
        const float sur   = 1.f / (1.f + __expf(-(snorm - thr)));
        s_mu[m] = smun; s_sig[m] = ssign;
        surp[m] = sur;
        avg_s[m] = 0.95f * avg_s[m] + 0.05f * sur;
      }
    }
    __syncthreads();

    // ---- Phase B3: err_proj / x_proj = (16x512)@(512x16), dual-acc WMMA ----
    if (wid < 2) {
      const bf16_t* src = (wid == 0) ? errbh : xbh;
      float*        dst = (wid == 0) ? eproj : xproj;
      v8f acc0 = {}, acc1 = {};
#pragma unroll 1
      for (int kt = 0; kt < 16; kt += 2) {
        acc0 = wmma_bf16(frag_A_bf(src + kt * 32, D_DIM, lane),
                         frag_B_bf(v_bf + kt * 32, D_DIM, lane), acc0);
        acc1 = wmma_bf16(frag_A_bf(src + (kt + 1) * 32, D_DIM, lane),
                         frag_B_bf(v_bf + (kt + 1) * 32, D_DIM, lane), acc1);
      }
      acc_to_lds(dst, R_DIM, acc0 + acc1, lane);
    }
    __syncthreads();

    // ---- Phase B4: Hebbian U update + Frobenius partial (H slice) ----
    // thread-per-(m,r), scalar over hh: benign 2-way LDS bank conflict
    {
      const int m = tid >> 4, r = tid & 15;
      const float pe = 0.1f * surp[m];          // BASE_PLASTICITY * surprise
      const float ep = eproj[m * R_DIM + r];
      float facc = 0.f;
#pragma unroll 4
      for (int hh = 0; hh < HS; ++hh) {
        const int ui = (m * HS + hh) * R_DIM + r;
        const float dU = pe * hb[m * HS + hh] * ep;
        const float un = Ubuf[ui] + 0.05f * (dU - 0.1f * (Ubuf[ui] - Utgt[ui]));
        Ubuf[ui] = un;                          // unscaled until global norm known
        facc += un * un;
      }
#pragma unroll
      for (int msk = 1; msk < 16; msk <<= 1) facc += __shfl_xor(facc, msk, 16);
      if (r == 0) fro_mine[m] = facc;
    }
    ++epoch; cl_signal(ctr);   // split barrier: signal, overlap GEMMs below

    // ---- Phase B5: big GEMMs hide the cluster barrier latency ----
    {
      const bool isBase = (wid < 4);
      const int nt = isBase ? wid : (wid - 4);
      const bf16_t* src = isBase ? xbh : errbh;
      const bf16_t* Wm  = (isBase ? wi_bf : we_bf) + (size_t)(h0 + nt * 16) * D_DIM;
      float* dst        = isBase ? baseb : errwb;
      v8f acc0 = {}, acc1 = {};
#pragma unroll 1
      for (int kt = 0; kt < 16; kt += 2) {
        acc0 = wmma_bf16(frag_A_bf(src + kt * 32, D_DIM, lane),
                         frag_B_bf(Wm + kt * 32, D_DIM, lane), acc0);
        acc1 = wmma_bf16(frag_A_bf(src + (kt + 1) * 32, D_DIM, lane),
                         frag_B_bf(Wm + (kt + 1) * 32, D_DIM, lane), acc1);
      }
      acc_to_lds(dst + nt * 16, HS, acc0 + acc1, lane);
    }
    cl_wait(ctr, epoch * NWG_CL);

    // ---- Phase B6: global Frobenius norm -> scale factor ----
    if (tid < MB) {
      float s = 0.f;
#pragma unroll
      for (int w = 0; w < NWG_CL; ++w) s += fro_base[w * MB + tid];
      const float fro = sqrtf(s);
      scl[tid] = fminf(1.f, 1.f / (fro + 1e-6f));   // TARGET_NORM = 1
    }
    __syncthreads();

    // ---- Phase B7a: apply scale; sleep-consolidate U_tgt ----
    {
      const int m = tid >> 4, r = tid & 15;
      const float sc = scl[m];
      const bool trig = (avg_s[m] < 0.3f);
#pragma unroll 4
      for (int hh = 0; hh < HS; ++hh) {
        const int ui = (m * HS + hh) * R_DIM + r;
        const float un = Ubuf[ui] * sc;
        Ubuf[ui] = un;
        if (trig) Utgt[ui] += 0.01f * (un - Utgt[ui]);
      }
    }
    __syncthreads();

    // ---- Phase B7b: fast-weight drive (vectorized), LTC update, emit h ----
    for (int idx = tid; idx < MB * HS; idx += NTHREADS) {
      const int m = idx >> 6, hh = idx & (HS - 1);
      float fe = 0.f;
#pragma unroll
      for (int q = 0; q < 4; ++q) {
        const float4 uv = *(const float4*)&Ubuf[idx * R_DIM + 4 * q];
        const float4 xq = *(const float4*)&xproj[m * R_DIM + 4 * q];
        fe += uv.x * xq.x + uv.y * xq.y + uv.z * xq.z + uv.w * xq.w;
      }
      const float sur = surp[m];
      const float ue  = (1.f + sur) * baseb[idx] + 0.1f * fe;
      const float hv  = hb[idx];
      const float hl  = hv + 0.05f * (-hv + tanhf(ue)) * (1.f + sur);
      const float hn  = (hl + errwb[idx]) * 0.99f + hv * 0.01f;
      hb[idx]  = hn;
      hbh[idx] = (bf16_t)hn;
      out[((size_t)(bbase + m) * T_STEPS + t) * H_DIM + h0 + hh] = hn;
    }
    // ---- Phase B7c: error running stats (vectorized, conflict-free) ----
    for (int i4 = tid * 4; i4 < MB * D_DIM; i4 += NTHREADS * 4) {
      const float4 e  = *(const float4*)&errb[i4];
      float4 mn = *(const float4*)&emean[i4];
      mn.x = 0.9f * mn.x + 0.1f * e.x;  mn.y = 0.9f * mn.y + 0.1f * e.y;
      mn.z = 0.9f * mn.z + 0.1f * e.z;  mn.w = 0.9f * mn.w + 0.1f * e.w;
      *(float4*)&emean[i4] = mn;
      float4 vv = *(const float4*)&evar[i4];
      const float dx = e.x - mn.x, dy = e.y - mn.y, dz = e.z - mn.z, dw = e.w - mn.w;
      vv.x = 0.9f * vv.x + 0.1f * dx * dx;  vv.y = 0.9f * vv.y + 0.1f * dy * dy;
      vv.z = 0.9f * vv.z + 0.1f * dz * dz;  vv.w = 0.9f * vv.w + 0.1f * dw * dw;
      *(float4*)&evar[i4] = vv;
    }
    __syncthreads();
  }
}

extern "C" void kernel_launch(void* const* d_in, const int* in_sizes, int n_in,
                              void* d_out, int out_size, void* d_ws, size_t ws_size,
                              hipStream_t stream) {
  (void)in_sizes; (void)n_in; (void)out_size; (void)ws_size;
  const float* x_seq  = (const float*)d_in[0];
  const float* W_pred = (const float*)d_in[1];
  const float* W_in   = (const float*)d_in[2];
  const float* W_err  = (const float*)d_in[3];
  const float* Vm     = (const float*)d_in[4];
  float* out = (float*)d_out;

  // workspace layout (~5.1 MB):
  //   [0,4K)       barrier counters (zeroed every call -> deterministic)
  //   [4K, +2M)    x_pred partials (scrambled layout), double buffered
  //   [+4K)        frobenius partials
  //   [+3M+16K)    bf16 weight copies (one-time in-kernel conversion)
  char* ws = (char*)d_ws;
  unsigned* ctr = (unsigned*)ws;
  float* xp  = (float*)(ws + 4096);
  size_t off = 4096 + (size_t)2 * N_CL * NWG_CL * MB * D_DIM * sizeof(float);
  float* fro = (float*)(ws + off);
  off += 4096;
  bf16_t* wpb = (bf16_t*)(ws + off); off += (size_t)D_DIM * H_DIM * sizeof(bf16_t);
  bf16_t* wib = (bf16_t*)(ws + off); off += (size_t)D_DIM * H_DIM * sizeof(bf16_t);
  bf16_t* web = (bf16_t*)(ws + off); off += (size_t)D_DIM * H_DIM * sizeof(bf16_t);
  bf16_t* vb  = (bf16_t*)(ws + off);

  hipMemsetAsync(ws, 0, 4096, stream);

  const size_t smem_bf  = (size_t)(2 * MB * D_DIM + MB * HS) * sizeof(bf16_t);
  const size_t smem_f32 = (size_t)(2 * MB * HS * R_DIM + 3 * MB * D_DIM + 3 * MB * HS +
                                   2 * MB * R_DIM + 5 * MB) * sizeof(float);
  const size_t smem = smem_bf + smem_f32;   // ~279 KB of 320 KB/WGP

  hipLaunchKernelGGL(dream_cell_kernel, dim3(N_CL * NWG_CL), dim3(NTHREADS), smem, stream,
                     x_seq, W_pred, W_in, W_err, Vm, out, ctr, xp, fro, wpb, wib, web, vb);
}
